// EdgeNavierStokesLayer_41128606827044
// MI455X (gfx1250) — matrix operand order repaired and checked by reading, not compiled, not run on hardware
//
#include <hip/hip_runtime.h>

#define N_NODES 50000
#define D_FEAT  128
#define N_EDGES 640000
#define DT_STEP 0.03f

typedef __bf16 bf16;
typedef __attribute__((ext_vector_type(4)))  __bf16 v4bf;
typedef __attribute__((ext_vector_type(8)))  __bf16 v8bf;
typedef __attribute__((ext_vector_type(16))) __bf16 v16bf;
typedef __attribute__((ext_vector_type(8)))  float  v8f;
typedef __attribute__((ext_vector_type(4)))  unsigned v4u;
typedef __attribute__((ext_vector_type(8)))  unsigned v8u;

#if __has_builtin(__builtin_amdgcn_tanhf)
#define FAST_TANH(x) __builtin_amdgcn_tanhf(x)
#elif __has_builtin(__builtin_amdgcn_tanh_f32)
#define FAST_TANH(x) __builtin_amdgcn_tanh_f32(x)
#else
#define FAST_TANH(x) tanhf(x)
#endif

// -------- weight blob layout (identical in d_ws and in LDS at offset 0) --------
#define OFF_VW1   0u        // 256x128 bf16 B-frags : 64KB
#define OFF_FW1   65536u    // 64KB
#define OFF_FW2   131072u   // 128x128 : 32KB
#define OFF_PW1   163840u   // 32KB
#define OFF_PW2   196608u   // 32KB
#define OFF_VW2F  229376u   // vw2 padded to 128x16 B-frags : 4KB
#define OFF_VB1   233472u   // f32[128]
#define OFF_FB1   233984u
#define OFF_FB2   234496u
#define OFF_PB1   235008u
#define OFF_PB2   235520u
#define OFF_VB2   236032u   // f32[1] (+pad)
#define WEIGHT_BYTES 236544u            // 8B-aligned; /8 = 29568 < 65536
// -------- per-pair tile region --------
#define OFF_TILES 236544u
#define TILE_STRIDE 16896u
#define T_X   0u            // 16x256 bf16 (concat hi|hj)
#define T_DH  8192u         // 16x128 bf16 (hi - hj)
#define T_I   12288u        // 16x128 bf16 intermediate (reused)
#define T_NU  16384u        // f32[16]
#define T_ROW 16448u        // int[16]
#define SMEM_BYTES (OFF_TILES + 4u * TILE_STRIDE)   // 304128 < 320KB

__device__ __forceinline__ v8f wmma_bf16(v16bf a, v16bf b, v8f c) {
  return __builtin_amdgcn_wmma_f32_16x16x32_bf16(false, a, false, b, (short)0, c,
                                                 false, false);
}

__device__ __forceinline__ v16bf cat8(v8bf lo, v8bf hi) {
  return __builtin_shufflevector(lo, hi, 0,1,2,3,4,5,6,7,8,9,10,11,12,13,14,15);
}

// A: 16xK bf16 row-major in LDS. slots0..15 -> K=32kt+16g+[0..15]
__device__ __forceinline__ v16bf load_A(const bf16* __restrict__ base, int rowStride,
                                        int lane, int kt) {
  const int M = lane & 15, g = lane >> 4;
  const bf16* p = base + M * rowStride + kt * 32 + g * 16;
  return cat8(*(const v8bf*)p, *(const v8bf*)(p + 8));
}

// B: fragment-swizzled, 32B contiguous per lane per 1KB fragment
__device__ __forceinline__ v16bf load_Bf(const char* __restrict__ wf, int fragIdx,
                                         int lane) {
  const bf16* p = (const bf16*)(wf + (fragIdx << 10)) + lane * 16;
  return cat8(*(const v8bf*)p, *(const v8bf*)(p + 8));
}

// ---- TDM: DMA the pre-swizzled weight blob (d_ws) into LDS[0..WEIGHT_BYTES) ----
__device__ __forceinline__ void tdm_load_weights(const char* ws) {
  const unsigned long long ga = (unsigned long long)(uintptr_t)ws;
  const unsigned n8 = WEIGHT_BYTES >> 3;          // elements of 8 bytes
  v4u g0;
  g0[0] = 1u;                                     // count=1, user descriptor
  g0[1] = 0u;                                     // lds_addr = 0
  g0[2] = (unsigned)ga;                           // global_addr[31:0]
  g0[3] = (unsigned)((ga >> 32) & 0x1FFFFFFu) | (2u << 30);  // addr[56:32] | type=2
  v8u g1;
  g1[0] = 3u << 16;                               // data_size=8B, no flags
  g1[1] = (n8 & 0xFFFFu) << 16;                   // tensor_dim0[15:0]
  g1[2] = (n8 >> 16) | (1u << 16);                // tensor_dim0[31:16] | tensor_dim1=1
  g1[3] = (n8 & 0xFFFFu) << 16;                   // tile_dim0
  g1[4] = 0u;                                     // tile_dim1/2 unused
  g1[5] = n8;                                     // tensor_dim0_stride lo
  g1[6] = 0u;
  g1[7] = 0u;
  asm volatile("tensor_load_to_lds %0, %1" :: "s"(g0), "s"(g1) : "memory");
}

// GEMM (M16 x N64-per-wave x K=32*KT) + bias + act -> activated bf16 Obuf (16x128)
template<int KT, int RELU>
__device__ __forceinline__ void gemm_act_store(const bf16* __restrict__ A, int rowStride,
                                               const char* __restrict__ Wf,
                                               const float* __restrict__ bias,
                                               bf16* __restrict__ Obuf,
                                               int wHalf, int lane) {
  const int g = lane >> 4;
  v8f z = {0.f,0.f,0.f,0.f,0.f,0.f,0.f,0.f};
  v8f acc[4] = {z, z, z, z};
#pragma unroll
  for (int kt = 0; kt < KT; ++kt) {
    v16bf a = load_A(A, rowStride, lane, kt);
#pragma unroll
    for (int nt4 = 0; nt4 < 4; ++nt4)
      acc[nt4] = wmma_bf16(a, load_Bf(Wf, (kt << 3) + (wHalf << 2) + nt4, lane),
                           acc[nt4]);
  }
#pragma unroll
  for (int nt4 = 0; nt4 < 4; ++nt4) {
    const int n = (lane & 15) + (((wHalf << 2) + nt4) << 4);
    const float bi = bias[n];
#pragma unroll
    for (int v = 0; v < 8; ++v) {
      float x = acc[nt4][v] + bi;
      x = RELU ? fmaxf(x, 0.0f) : FAST_TANH(x);
      Obuf[(v + (g << 3)) * 128 + n] = (bf16)x;
    }
  }
}

// Second-layer GEMM (K=128), accumulators stay in VGPRs
__device__ __forceinline__ void gemm2_acc(const bf16* __restrict__ A,
                                          const char* __restrict__ Wf,
                                          int wHalf, int lane, v8f acc[4]) {
  v8f z = {0.f,0.f,0.f,0.f,0.f,0.f,0.f,0.f};
#pragma unroll
  for (int nt4 = 0; nt4 < 4; ++nt4) acc[nt4] = z;
#pragma unroll
  for (int kt = 0; kt < 4; ++kt) {
    v16bf a = load_A(A, 128, lane, kt);
#pragma unroll
    for (int nt4 = 0; nt4 < 4; ++nt4)
      acc[nt4] = wmma_bf16(a, load_Bf(Wf, (kt << 3) + (wHalf << 2) + nt4, lane),
                           acc[nt4]);
  }
}

// ---------------- one-time weight prep: f32 -> bf16 fragment blob in d_ws ----------------
__device__ __forceinline__ void swizzle_region(char* dst, const float* __restrict__ W,
                                               int K, int t, int nT) {
  bf16* d = (bf16*)dst;
  const int total = K << 7;
  for (int i = t; i < total; i += nT) {
    const int k = i >> 7, n = i & 127;
    const int kt = k >> 5, kr = k & 31;
    const int nt = n >> 4, nr = n & 15;
    const int idx = (((kt << 3) + nt) << 9)
                  + ((nr + ((kr >> 4) << 4)) << 4) + (kr & 15);
    d[idx] = (bf16)W[i];
  }
}

__global__ void prep_weights_kernel(const float* __restrict__ vw1, const float* __restrict__ fw1,
                                    const float* __restrict__ fw2, const float* __restrict__ pw1,
                                    const float* __restrict__ pw2, const float* __restrict__ vw2,
                                    const float* __restrict__ vb1, const float* __restrict__ fb1,
                                    const float* __restrict__ fb2, const float* __restrict__ pb1,
                                    const float* __restrict__ pb2, const float* __restrict__ vb2,
                                    char* __restrict__ ws) {
  const int t  = blockIdx.x * 256 + threadIdx.x;
  const int nT = gridDim.x * 256;
  swizzle_region(ws + OFF_VW1, vw1, 256, t, nT);
  swizzle_region(ws + OFF_FW1, fw1, 256, t, nT);
  swizzle_region(ws + OFF_FW2, fw2, 128, t, nT);
  swizzle_region(ws + OFF_PW1, pw1, 128, t, nT);
  swizzle_region(ws + OFF_PW2, pw2, 128, t, nT);
  // vw2 (128x1) zero-padded to 128x16 B fragments (only column 0 nonzero)
  for (int i = t; i < 2048; i += nT) {
    const int kt = i >> 9, r = i & 511;
    const int l = r >> 4, s = r & 15;
    const int K = kt * 32 + ((l >> 4) << 4) + s;
    ((bf16*)(ws + OFF_VW2F))[i] = ((l & 15) == 0) ? (bf16)vw2[K] : (bf16)0.0f;
  }
  if (t < 128) {
    ((float*)(ws + OFF_VB1))[t] = vb1[t];
    ((float*)(ws + OFF_FB1))[t] = fb1[t];
    ((float*)(ws + OFF_FB2))[t] = fb2[t];
    ((float*)(ws + OFF_PB1))[t] = pb1[t];
    ((float*)(ws + OFF_PB2))[t] = pb2[t];
  }
  if (t == 0) *(float*)(ws + OFF_VB2) = vb2[0];
}

__global__ void init_out_kernel(const float4* __restrict__ h4, float4* __restrict__ o4) {
  const int i = blockIdx.x * 256 + threadIdx.x;   // 6250*256 == N*D/4 exactly
  o4[i] = h4[i];
}

__global__ __launch_bounds__(256)
void edge_ns_kernel(const float* __restrict__ h, const int* __restrict__ ei,
                    const char* __restrict__ ws, float* __restrict__ out) {
  extern __shared__ float4 smem4[];
  char* smem = (char*)smem4;

  const int tid   = threadIdx.x;
  const int lane  = tid & 31;
  const int wave  = tid >> 5;       // 0..7
  const int tile  = wave >> 1;      // 0..3 : 16-edge tile per wave pair
  const int wHalf = wave & 1;       // which 64 output columns this wave computes

  // ---- kick off TDM weight DMA (overlaps with the edge gather below) ----
  if (wave == 0) tdm_load_weights(ws);

  // ---- stage this pair's 16-edge tile: X=[hi|hj] bf16, Dh=hi-hj bf16 ----
  const int* rowp = ei;
  const int* colp = ei + N_EDGES;
  const int edge0 = blockIdx.x * 64 + tile * 16;

  char* tb   = smem + OFF_TILES + (unsigned)tile * TILE_STRIDE;
  bf16*  X   = (bf16*)(tb + T_X);
  bf16*  Dh  = (bf16*)(tb + T_DH);
  bf16*  Ibuf= (bf16*)(tb + T_I);
  float* nuB = (float*)(tb + T_NU);
  int*  rows = (int*)(tb + T_ROW);

  const int ptid = (wHalf << 5) + lane;        // 0..63 within pair
  if (ptid < 16) rows[ptid] = rowp[edge0 + ptid];
#pragma unroll
  for (int i = ptid; i < 16 * 32; i += 64) {
    const int e = i >> 5, q = i & 31;          // e uniform per iteration
    const int ri = rowp[edge0 + e];
    const int ci = colp[edge0 + e];
    const float4 a4 = *(const float4*)(h + ri * D_FEAT + q * 4);
    const float4 b4 = *(const float4*)(h + ci * D_FEAT + q * 4);
    v4bf xa = {(bf16)a4.x, (bf16)a4.y, (bf16)a4.z, (bf16)a4.w};
    v4bf xb = {(bf16)b4.x, (bf16)b4.y, (bf16)b4.z, (bf16)b4.w};
    v4bf xd = {(bf16)(a4.x - b4.x), (bf16)(a4.y - b4.y),
               (bf16)(a4.z - b4.z), (bf16)(a4.w - b4.w)};
    *(v4bf*)(X  + e * 256 + q * 4)       = xa;
    *(v4bf*)(X  + e * 256 + 128 + q * 4) = xb;
    *(v4bf*)(Dh + e * 128 + q * 4)       = xd;
  }
  if (wave == 0) __builtin_amdgcn_s_wait_tensorcnt(0);
  __syncthreads();

  // ---- V1 = tanh(X @ vw1 + vb1) -> Ibuf ----
  gemm_act_store<8, 0>(X, 256, smem + OFF_VW1, (const float*)(smem + OFF_VB1),
                       Ibuf, wHalf, lane);
  __syncthreads();

  // ---- nu = V1 @ vw2 + vb2 via WMMA against zero-padded B (col 0 only) ----
  if (wHalf == 0) {
    v8f acc = {0.f,0.f,0.f,0.f,0.f,0.f,0.f,0.f};
#pragma unroll
    for (int kt = 0; kt < 4; ++kt)
      acc = wmma_bf16(load_A(Ibuf, 128, lane, kt),
                      load_Bf(smem + OFF_VW2F, kt, lane), acc);
    if ((lane & 15) == 0) {
      const float vb2v = *(const float*)(smem + OFF_VB2);
#pragma unroll
      for (int v = 0; v < 8; ++v) nuB[v + ((lane >> 4) << 3)] = acc[v] + vb2v;
    }
  }
  __syncthreads();

  // ---- F1 = relu(X @ fw1 + fb1) -> Ibuf (V1 dead) ----
  gemm_act_store<8, 1>(X, 256, smem + OFF_FW1, (const float*)(smem + OFF_FB1),
                       Ibuf, wHalf, lane);
  __syncthreads();

  // ---- F2 = F1 @ fw2 (bias later), in VGPRs ----
  v8f accF[4];
  gemm2_acc(Ibuf, smem + OFF_FW2, wHalf, lane, accF);
  __syncthreads();

  // ---- P1 = tanh(Dh @ pw1 + pb1) -> Ibuf ----
  gemm_act_store<4, 0>(Dh, 128, smem + OFF_PW1, (const float*)(smem + OFF_PB1),
                       Ibuf, wHalf, lane);
  __syncthreads();

  // ---- P2 = P1 @ pw2 (bias later), in VGPRs ----
  v8f accP[4];
  gemm2_acc(Ibuf, smem + OFF_PW2, wHalf, lane, accP);

  // ---- message = nu*(hj-hi) + F2 - P2 ; scatter-add DT*message ----
  const float* fb2s = (const float*)(smem + OFF_FB2);
  const float* pb2s = (const float*)(smem + OFF_PB2);
  const int g = lane >> 4;
#pragma unroll
  for (int nt4 = 0; nt4 < 4; ++nt4) {
    const int n = (lane & 15) + (((wHalf << 2) + nt4) << 4);
    const float fb = fb2s[n];
    const float pb = pb2s[n];
#pragma unroll
    for (int v = 0; v < 8; ++v) {
      const int e = v + (g << 3);
      const float dh = (float)Dh[e * 128 + n];          // hi - hj
      const float msg = (accF[nt4][v] + fb) - (accP[nt4][v] + pb) - nuB[e] * dh;
      atomicAdd(out + rows[e] * D_FEAT + n, DT_STEP * msg);
    }
  }
}

extern "C" void kernel_launch(void* const* d_in, const int* in_sizes, int n_in,
                              void* d_out, int out_size, void* d_ws, size_t ws_size,
                              hipStream_t stream) {
  const float* h   = (const float*)d_in[0];
  const int*   ei  = (const int*)d_in[1];
  const float* vw1 = (const float*)d_in[2];
  const float* vb1 = (const float*)d_in[3];
  const float* vw2 = (const float*)d_in[4];
  const float* vb2 = (const float*)d_in[5];
  const float* pw1 = (const float*)d_in[6];
  const float* pb1 = (const float*)d_in[7];
  const float* pw2 = (const float*)d_in[8];
  const float* pb2 = (const float*)d_in[9];
  const float* fw1 = (const float*)d_in[10];
  const float* fb1 = (const float*)d_in[11];
  const float* fw2 = (const float*)d_in[12];
  const float* fb2 = (const float*)d_in[13];
  float* out = (float*)d_out;
  char*  ws  = (char*)d_ws;

  // 1) one-time weight convert+swizzle into d_ws (bf16 WMMA-B fragment blob)
  hipLaunchKernelGGL(prep_weights_kernel, dim3(64), dim3(256), 0, stream,
                     vw1, fw1, fw2, pw1, pw2, vw2, vb1, fb1, fb2, pb1, pb2, vb2, ws);

  // 2) out = h (atomics then accumulate DT * message)
  hipLaunchKernelGGL(init_out_kernel, dim3((N_NODES * D_FEAT) / 1024), dim3(256), 0,
                     stream, (const float4*)h, (float4*)out);

  // 3) edge GEMMs: 64 edges per block -> exactly 10000 blocks
  hipLaunchKernelGGL(edge_ns_kernel, dim3(N_EDGES / 64), dim3(256), SMEM_BYTES, stream,
                     h, ei, ws, out);
}
